// MultiHeadAttentionBlock_72816875536537
// MI455X (gfx1250) — matrix operand order repaired and checked by reading, not compiled, not run on hardware
//
#include <hip/hip_runtime.h>
#include <hip/hip_bf16.h>

// MI455X (gfx1250) MHA block: bf16 WMMA GEMMs (TDM-staged, double-buffered)
// + flash-style causal attention.  B=2, S=2048, D=1024, H=16, HD=64.

typedef __bf16 bf16_t;
typedef __bf16 v16bf __attribute__((ext_vector_type(16)));
typedef __bf16 v8bf  __attribute__((ext_vector_type(8)));
typedef __bf16 v4bf  __attribute__((ext_vector_type(4)));
typedef float  v8f   __attribute__((ext_vector_type(8)));
typedef float  v4f   __attribute__((ext_vector_type(4)));

#define B_  2
#define S_  2048
#define D_  1024
#define H_  16
#define HD_ 64

#if defined(__AMDGCN__) && __has_builtin(__builtin_amdgcn_tensor_load_to_lds) && \
    __has_builtin(__builtin_amdgcn_s_wait_tensorcnt)
#define USE_TDM 1
#else
#define USE_TDM 0
#endif

static __device__ __forceinline__ v16bf cat8(v8bf a, v8bf b) {
  return __builtin_shufflevector(a, b, 0,1,2,3,4,5,6,7,8,9,10,11,12,13,14,15);
}

#if USE_TDM
typedef unsigned int v4u __attribute__((ext_vector_type(4)));
typedef int          v8i __attribute__((ext_vector_type(8)));
typedef int          v4i __attribute__((ext_vector_type(4)));

// Issue a 2-D TDM tile load (bf16 elements) global -> LDS, with LDS row padding.
// D# bit layout per CDNA5 ISA 8.3/8.4: group0 = {flags, lds_addr, global_addr,
// type=2}; group1 = {data_size/pad ctl, tensor dims, tile dims, stride}.
static __device__ __forceinline__ void tdm_load_2d_bf16(
    const bf16_t* gptr, unsigned ldsByteOff,
    unsigned tensor_d0, unsigned tensor_d1, unsigned stride0_elems,
    unsigned tile_d0, unsigned tile_d1,
    unsigned pad_interval_code, unsigned pad_amount_code, bool pad_en)
{
  const unsigned long long ga = (unsigned long long)(uintptr_t)gptr;
  v4u g0;
  g0.x = 1u;                                   // count=1, user descriptor
  g0.y = ldsByteOff;                           // LDS byte address
  g0.z = (unsigned)(ga & 0xFFFFFFFFu);         // global_addr[31:0]
  g0.w = (unsigned)((ga >> 32) & 0x01FFFFFFu)  // global_addr[56:32]
         | 0x80000000u;                        // type=2 ("image")
  v8i g1;
  g1[0] = (int)((1u << 16)                     // data_size = 2 bytes
                | (pad_en ? (1u << 20) : 0u)   // pad_enable
                | (pad_interval_code << 22)    // DWORDs before pad (3 -> 16)
                | (pad_amount_code << 25));    // pad DWORDs (7 -> 8)
  g1[1] = (int)((tensor_d0 & 0xFFFFu) << 16);                                  // dim0[15:0]
  g1[2] = (int)(((tensor_d0 >> 16) & 0xFFFFu) | ((tensor_d1 & 0xFFFFu) << 16)); // dim0[31:16],dim1[15:0]
  g1[3] = (int)(((tensor_d1 >> 16) & 0xFFFFu) | ((tile_d0 & 0xFFFFu) << 16));   // dim1[31:16],tile0
  g1[4] = (int)(tile_d1 & 0xFFFFu);            // tile1 (tile2 = 0)
  g1[5] = (int)stride0_elems;                  // tensor_dim0_stride[31:0]
  g1[6] = 0;                                   // stride hi + dim1_stride lo
  g1[7] = 0;
  const v4i z4 = {0, 0, 0, 0};
#if __clang_major__ >= 23
  const v8i z8 = {0, 0, 0, 0, 0, 0, 0, 0};
  __builtin_amdgcn_tensor_load_to_lds(g0, g1, z4, z4, z8, 0);
#else
  __builtin_amdgcn_tensor_load_to_lds(g0, g1, z4, z4, 0);
#endif
}
#endif  // USE_TDM

// ---------------------------------------------------------------- cast kernel
__global__ __launch_bounds__(256)
void cast_f32_to_bf16(const float* __restrict__ src, bf16_t* __restrict__ dst, int n4) {
  int i = blockIdx.x * 256 + threadIdx.x;
  if (i < n4) {
    v4f x = ((const v4f*)src)[i];
    v4bf y;
    for (int j = 0; j < 4; ++j) y[j] = (bf16_t)x[j];
    ((v4bf*)dst)[i] = y;
  }
}

// ------------------------------------------------------------- WMMA GEMM
// C[M,N] = A[M,K] * W[N,K]^T   (both A and W are K-contiguous row-major)
// MODE 0: store bf16 scattered to [B,H,S,HD] head layout (projections)
// MODE 1: store fp32 row-major [M,N] (output projection)
// Staging: TDM double-buffered pipeline (wave 0 issues tensor_load_to_lds for
// the next K-slab, waits TENSORcnt<=2 for the current one); fallback = manual
// vectorized global->LDS copies.
template<int MODE>
__global__ __launch_bounds__(256)
void gemm_bf16_wmma(const bf16_t* __restrict__ A, const bf16_t* __restrict__ W,
                    float* __restrict__ Cf, bf16_t* __restrict__ Cb,
                    int M, int N, int K)
{
#if USE_TDM
  __shared__ bf16_t As[2][128][48];   // 32-elt K-slab + 16-elt pad (row = 96B)
  __shared__ bf16_t Bs[2][128][48];
#else
  __shared__ bf16_t As[1][128][48];
  __shared__ bf16_t Bs[1][128][48];
#endif

  const int m0 = blockIdx.x * 128;
  const int n0 = blockIdx.y * 128;
  const int t  = threadIdx.x;
  const int wv = t >> 5, lane = t & 31;
  const int lr = lane & 15, lh = lane >> 4;
  const int wm = wv >> 2, wn = wv & 3;      // waves: 2 (M) x 4 (N)

  v8f acc[4][2];
  {
    v8f z = {};
    for (int i = 0; i < 4; ++i) for (int j = 0; j < 2; ++j) acc[i][j] = z;
  }

  const int nSlab = K / 32;

#if USE_TDM
  if (wv == 0) {   // prologue: slab 0 into buffer 0 (2 tensor ops in flight)
    tdm_load_2d_bf16(&A[(size_t)m0 * K], (unsigned)(uintptr_t)&As[0][0][0],
                     (unsigned)K, (unsigned)M, (unsigned)K, 32u, 128u, 3u, 7u, true);
    tdm_load_2d_bf16(&W[(size_t)n0 * K], (unsigned)(uintptr_t)&Bs[0][0][0],
                     (unsigned)K, (unsigned)N, (unsigned)K, 32u, 128u, 3u, 7u, true);
  }
  for (int s = 0; s < nSlab; ++s) {
    const int cur = s & 1, nxt = cur ^ 1;
    __syncthreads();    // all waves done reading buffer `nxt` (from slab s-1)
    if (wv == 0) {
      if (s + 1 < nSlab) {
        const int k1 = (s + 1) * 32;
        tdm_load_2d_bf16(&A[(size_t)m0 * K + k1], (unsigned)(uintptr_t)&As[nxt][0][0],
                         (unsigned)K, (unsigned)M, (unsigned)K, 32u, 128u, 3u, 7u, true);
        tdm_load_2d_bf16(&W[(size_t)n0 * K + k1], (unsigned)(uintptr_t)&Bs[nxt][0][0],
                         (unsigned)K, (unsigned)N, (unsigned)K, 32u, 128u, 3u, 7u, true);
        __builtin_amdgcn_s_wait_tensorcnt(2);  // slab s complete, s+1 in flight
      } else {
        __builtin_amdgcn_s_wait_tensorcnt(0);
      }
    }
    __syncthreads();    // slab s visible in LDS to all waves
#else
  const int srow = t >> 1, sh = t & 1;      // staging: 32B per thread per tile
  for (int s = 0; s < nSlab; ++s) {
    const int cur = 0;
    const int k0 = s * 32;
    __syncthreads();
    *(v16bf*)&As[0][srow][sh * 16] = *(const v16bf*)&A[(size_t)(m0 + srow) * K + k0 + sh * 16];
    *(v16bf*)&Bs[0][srow][sh * 16] = *(const v16bf*)&W[(size_t)(n0 + srow) * K + k0 + sh * 16];
    if (k0 + 32 < K) {
      __builtin_prefetch(&A[(size_t)(m0 + srow) * K + k0 + 32 + sh * 16], 0, 0);
      __builtin_prefetch(&W[(size_t)(n0 + srow) * K + k0 + 32 + sh * 16], 0, 0);
    }
    __syncthreads();
#endif

    v16bf bfrag[2];
    for (int nt = 0; nt < 2; ++nt)
      bfrag[nt] = *(const v16bf*)&Bs[cur][wn * 32 + nt * 16 + lr][lh * 16];

    for (int mt = 0; mt < 4; ++mt) {
      v8bf a0 = *(const v8bf*)&As[cur][wm * 64 + mt * 16 + lr][lh * 16];
      v8bf a1 = *(const v8bf*)&As[cur][wm * 64 + mt * 16 + lr][lh * 16 + 8];
      v16bf afrag = cat8(a0, a1);
      for (int nt = 0; nt < 2; ++nt)
        acc[mt][nt] = __builtin_amdgcn_wmma_f32_16x16x32_bf16(
            false, afrag, false, bfrag[nt], (short)0, acc[mt][nt], false, false);
    }
  }

  for (int mt = 0; mt < 4; ++mt) {
    for (int nt = 0; nt < 2; ++nt) {
      const int mrow = m0 + wm * 64 + mt * 16 + lh * 8;
      const int ncol = n0 + wn * 32 + nt * 16 + lr;
      if (MODE == 0) {
        const int h = ncol >> 6, hd = ncol & 63;
        for (int j = 0; j < 8; ++j) {
          const int m = mrow + j;
          const int b = m >> 11, s = m & 2047;
          Cb[(((size_t)(b * H_ + h)) * S_ + s) * HD_ + hd] = (bf16_t)acc[mt][nt][j];
        }
      } else {
        for (int j = 0; j < 8; ++j)
          Cf[(size_t)(mrow + j) * N + ncol] = acc[mt][nt][j];
      }
    }
  }
}

// ------------------------------------------------------- flash attention
// qp/kp/vp: bf16 [B,H,S,HD]. ctx out: bf16 [B,S,D]. Causal, online softmax.
__global__ __launch_bounds__(256)
void attention_wmma(const bf16_t* __restrict__ qp, const bf16_t* __restrict__ kp,
                    const bf16_t* __restrict__ vp, bf16_t* __restrict__ ctx)
{
  __shared__ bf16_t Kt[64][64];        // K tile, row-major == B of Q*K^T
  __shared__ bf16_t Vt[64][64];        // V tile transposed == B of P*V
  __shared__ bf16_t Ps[8][16][64];     // per-wave P scratch (C->A relayout)

  const int qt = blockIdx.x;           // q tile of 128 rows
  const int bh = blockIdx.y;           // b*H + h
  const int b  = bh >> 4, h = bh & 15;
  const int t  = threadIdx.x;
  const int wv = t >> 5, lane = t & 31;
  const int lr = lane & 15, lh = lane >> 4;

  const size_t headOff = (size_t)bh * S_ * HD_;
  const bf16_t* qh = qp + headOff;
  const bf16_t* kh = kp + headOff;
  const bf16_t* vh = vp + headOff;

  const int qRow0 = qt * 128 + wv * 16;   // wave owns 16 q rows

  // Q fragments (A layout: lane holds row lr; half lh holds k = 16*lh + s of each 32-chunk)
  v16bf qf[2];
  {
    const bf16_t* qrow = qh + (size_t)(qRow0 + lr) * HD_;
    for (int c = 0; c < 2; ++c) {
      v8bf a0 = *(const v8bf*)(qrow + c * 32 + lh * 16);
      v8bf a1 = *(const v8bf*)(qrow + c * 32 + lh * 16 + 8);
      qf[c] = cat8(a0, a1);
    }
  }

  v8f oacc[4];
  { v8f z = {}; for (int i = 0; i < 4; ++i) oacc[i] = z; }
  float mrow[8], lrow[8];
  for (int j = 0; j < 8; ++j) { mrow[j] = -1.0e30f; lrow[j] = 0.0f; }

  const int nkv = qt * 2 + 2;             // kv tiles of 64 (causal bound)
  const int srow = t >> 2, schk = t & 3;  // staging assignment

  for (int kt = 0; kt < nkv; ++kt) {
    __syncthreads();
    {
      const size_t base = (size_t)(kt * 64 + srow) * HD_ + schk * 16;
      *(v16bf*)&Kt[srow][schk * 16] = *(const v16bf*)(kh + base);
      v16bf vvv = *(const v16bf*)(vh + base);
      for (int e = 0; e < 16; ++e) Vt[schk * 16 + e][srow] = vvv[e];
    }
    __syncthreads();

    // ---- S = Q K^T (8 WMMAs: 4 n-tiles x 2 K-chunks)
    v8f sc[4];
    { v8f z = {}; for (int i = 0; i < 4; ++i) sc[i] = z; }
    for (int nt = 0; nt < 4; ++nt)
      for (int c = 0; c < 2; ++c) {
        v16bf bfrag = *(const v16bf*)&Kt[nt * 16 + lr][c * 32 + lh * 16];
        sc[nt] = __builtin_amdgcn_wmma_f32_16x16x32_bf16(
            false, qf[c], false, bfrag, (short)0, sc[nt], false, false);
      }

    // ---- online softmax (row spread across 16 lanes of a half-wave)
    const float scale = 0.125f;   // 1/sqrt(64)
    float p[4][8];
    for (int j = 0; j < 8; ++j) {
      const int rg = qRow0 + lh * 8 + j;
      float mx = -1.0e30f;
      for (int nt = 0; nt < 4; ++nt) {
        const int cg = kt * 64 + nt * 16 + lr;
        const float sv = (cg <= rg) ? (float)sc[nt][j] * scale : -__builtin_inff();
        p[nt][j] = sv;
        mx = fmaxf(mx, sv);
      }
      for (int off = 1; off < 16; off <<= 1) mx = fmaxf(mx, __shfl_xor(mx, off, 32));
      const float mnew  = fmaxf(mrow[j], mx);
      const float alpha = __expf(mrow[j] - mnew);
      float sum = 0.0f;
      for (int nt = 0; nt < 4; ++nt) {
        const float e = __expf(p[nt][j] - mnew);
        p[nt][j] = e;
        sum += e;
      }
      for (int off = 1; off < 16; off <<= 1) sum += __shfl_xor(sum, off, 32);
      lrow[j] = lrow[j] * alpha + sum;
      mrow[j] = mnew;
      for (int dt = 0; dt < 4; ++dt) oacc[dt][j] *= alpha;
      for (int nt = 0; nt < 4; ++nt)
        Ps[wv][lh * 8 + j][nt * 16 + lr] = (bf16_t)p[nt][j];
    }
    __syncthreads();   // order Ps stores before A-layout reload

    // ---- O += P V (8 WMMAs: 2 K-chunks x 4 d-tiles)
    for (int c = 0; c < 2; ++c) {
      v8bf a0 = *(const v8bf*)&Ps[wv][lr][c * 32 + lh * 16];
      v8bf a1 = *(const v8bf*)&Ps[wv][lr][c * 32 + lh * 16 + 8];
      v16bf pf = cat8(a0, a1);
      for (int dt = 0; dt < 4; ++dt) {
        v16bf bfrag = *(const v16bf*)&Vt[dt * 16 + lr][c * 32 + lh * 16];
        oacc[dt] = __builtin_amdgcn_wmma_f32_16x16x32_bf16(
            false, pf, false, bfrag, (short)0, oacc[dt], false, false);
      }
    }
  }

  // ---- epilogue: normalize, store ctx bf16 [B,S,D]
  for (int dt = 0; dt < 4; ++dt) {
    for (int j = 0; j < 8; ++j) {
      const int rg   = qRow0 + lh * 8 + j;
      const int dcol = h * 64 + dt * 16 + lr;
      const float o  = oacc[dt][j] / lrow[j];
      ctx[((size_t)b * S_ + rg) * D_ + dcol] = (bf16_t)o;
    }
  }
}

// ---------------------------------------------------------------- launcher
extern "C" void kernel_launch(void* const* d_in, const int* in_sizes, int n_in,
                              void* d_out, int out_size, void* d_ws, size_t ws_size,
                              hipStream_t stream) {
  const float* q  = (const float*)d_in[0];
  const float* k  = (const float*)d_in[1];
  const float* v  = (const float*)d_in[2];
  // d_in[3] = boolean causal mask: applied analytically in-kernel
  const float* Wq = (const float*)d_in[4];
  const float* Wk = (const float*)d_in[5];
  const float* Wv = (const float*)d_in[6];
  const float* Wo = (const float*)d_in[7];
  float* out = (float*)d_out;

  const size_t MS = (size_t)B_ * S_ * D_;  // 4,194,304
  const size_t WS = (size_t)D_ * D_;       // 1,048,576

  bf16_t* qb  = (bf16_t*)d_ws;
  bf16_t* kb  = qb  + MS;
  bf16_t* vb  = kb  + MS;
  bf16_t* wqb = vb  + MS;
  bf16_t* wkb = wqb + WS;
  bf16_t* wvb = wkb + WS;
  bf16_t* wob = wvb + WS;
  bf16_t* qpb = wob + WS;
  bf16_t* kpb = qpb + MS;
  bf16_t* vpb = kpb + MS;
  bf16_t* ctx = vpb + MS;

  auto cast = [&](const float* s, bf16_t* d, size_t n) {
    const int n4 = (int)(n / 4);
    cast_f32_to_bf16<<<(n4 + 255) / 256, 256, 0, stream>>>(s, d, n4);
  };
  cast(q, qb, MS);  cast(k, kb, MS);  cast(v, vb, MS);
  cast(Wq, wqb, WS); cast(Wk, wkb, WS); cast(Wv, wvb, WS); cast(Wo, wob, WS);

  const dim3 gg(B_ * S_ / 128, D_ / 128);   // 32 x 8
  gemm_bf16_wmma<0><<<gg, 256, 0, stream>>>(qb, wqb, nullptr, qpb, B_ * S_, D_, D_);
  gemm_bf16_wmma<0><<<gg, 256, 0, stream>>>(kb, wkb, nullptr, kpb, B_ * S_, D_, D_);
  gemm_bf16_wmma<0><<<gg, 256, 0, stream>>>(vb, wvb, nullptr, vpb, B_ * S_, D_, D_);

  attention_wmma<<<dim3(S_ / 128, B_ * H_), 256, 0, stream>>>(qpb, kpb, vpb, ctx);

  gemm_bf16_wmma<1><<<gg, 256, 0, stream>>>(ctx, wob, out, nullptr, B_ * S_, D_, D_);
}
